// MaskedMultiHeadedSelfAttention_49486613185051
// MI455X (gfx1250) — matrix operand order repaired and checked by reading, not compiled, not run on hardware
//
#include <hip/hip_runtime.h>
#include <stdint.h>

// ---------------------------------------------------------------------------
// Problem constants (fixed by the reference)
// ---------------------------------------------------------------------------
constexpr int Bc  = 2;
constexpr int Sc  = 2048;    // power of two: use >>11 / &2047
constexpr int Dc  = 1024;
constexpr int Hc  = 16;
constexpr int HDc = 64;      // head dim

// ---------------------------------------------------------------------------
// Types / helpers
// ---------------------------------------------------------------------------
typedef __attribute__((ext_vector_type(16))) __bf16        v16bf;
typedef __attribute__((ext_vector_type(8)))  float         v8f;
typedef __attribute__((ext_vector_type(4)))  float         f32x4;
typedef __attribute__((ext_vector_type(4)))  unsigned int  u32x4;
typedef __attribute__((ext_vector_type(2)))  unsigned int  u32x2;
typedef __attribute__((ext_vector_type(4)))  int           i32x4;

union Frag16 { v16bf v; u32x4 u[2]; };

__device__ __forceinline__ unsigned short f2bf(float f) {
  union { float f; unsigned int u; } cv; cv.f = f;
  unsigned int r = cv.u + 0x7FFFu + ((cv.u >> 16) & 1u);  // round-nearest-even
  return (unsigned short)(r >> 16);
}

// packed pair conversion: prefer the HW packed cvt if the builtin exists
__device__ __forceinline__ unsigned int pk_bf16(float lo, float hi) {
#if __has_builtin(__builtin_amdgcn_cvt_pk_bf16_f32)
  typedef __attribute__((ext_vector_type(2))) __bf16 v2bf;
  union { v2bf v; unsigned int u; } cv;
  cv.v = __builtin_amdgcn_cvt_pk_bf16_f32(lo, hi);
  return cv.u;
#else
  return (unsigned int)f2bf(lo) | ((unsigned int)f2bf(hi) << 16);
#endif
}

// 16-byte global -> LDS copy; async (ASYNCcnt) when available
#if __has_builtin(__builtin_amdgcn_global_load_async_to_lds_b128)
#define HAS_ASYNC_LDS 1
typedef i32x4 __attribute__((address_space(1))) * gi32x4_p;  // global int4*
typedef i32x4 __attribute__((address_space(3))) * li32x4_p;  // LDS int4*
#else
#define HAS_ASYNC_LDS 0
#endif

__device__ __forceinline__ void copy16_g2l(unsigned short* lds,
                                           const unsigned short* g) {
#if HAS_ASYNC_LDS
  __builtin_amdgcn_global_load_async_to_lds_b128(
      (gi32x4_p)(g), (li32x4_p)(lds), 0, 0);
#else
  *(u32x4*)lds = *(const u32x4*)g;
#endif
}
__device__ __forceinline__ void async_join() {
#if HAS_ASYNC_LDS
  asm volatile("s_wait_asynccnt 0" ::: "memory");
#endif
}

// ---------------------------------------------------------------------------
// Kernel 1: fused per-head Q/K/V projection  (bf16 WMMA, fp32 accumulate)
// grid = (BS/128, H, 3); block = 256 (8 waves); wave -> 16 rows x 64 cols
// V (z==2) is stored TRANSPOSED [b][h][e][s] so attention can stage it with
// contiguous 16B copies.
// ---------------------------------------------------------------------------
__global__ __launch_bounds__(256)
void qkv_proj_kernel(const float* __restrict__ x,
                     const float* __restrict__ Wq, const float* __restrict__ bq,
                     const float* __restrict__ Wk, const float* __restrict__ bk,
                     const float* __restrict__ Wv, const float* __restrict__ bv,
                     unsigned short* __restrict__ Qb,
                     unsigned short* __restrict__ Kb,
                     unsigned short* __restrict__ Vb)
{
  __shared__ unsigned short Xs[128 * 64];   // A tile [row][k], 16 KB
  __shared__ unsigned short Wt[64 * 64];    // B tile TRANSPOSED [n][k], 8 KB

  const int tid  = threadIdx.x;
  const int lane = tid & 31;
  const int wave = tid >> 5;
  const int row0 = blockIdx.x * 128;
  const int h    = blockIdx.y;
  const int z    = blockIdx.z;

  const float*    W    = (z == 0) ? Wq : ((z == 1) ? Wk : Wv);
  const float*    bias = (z == 0) ? bq : ((z == 1) ? bk : bv);
  unsigned short* Out  = (z == 0) ? Qb : ((z == 1) ? Kb : Vb);

  v8f acc[4] = {};
  const int half = lane >> 4;
  const int arow = wave * 16 + (lane & 15);

  for (int kb = 0; kb < Dc / 64; ++kb) {
    __syncthreads();
    // stage x tile 128x64 (float4 load, packed cvt, b64 LDS store)
    for (int i = tid; i < 2048; i += 256) {
      const int r = i >> 4, c4 = (i & 15) * 4;
      f32x4 xv = *(const f32x4*)&x[(size_t)(row0 + r) * Dc + kb * 64 + c4];
      u32x2 p; p[0] = pk_bf16(xv[0], xv[1]); p[1] = pk_bf16(xv[2], xv[3]);
      *(u32x2*)&Xs[r * 64 + c4] = p;
    }
    // stage W tile transposed: Wt[n][k], 4 k per thread (coalesced over n)
    for (int i = tid; i < 1024; i += 256) {
      const int n = i & 63, k0 = (i >> 6) * 4;
      const float* wp = &W[((size_t)h * Dc + kb * 64 + k0) * HDc + n];
      u32x2 p;
      p[0] = pk_bf16(wp[0], wp[HDc]);
      p[1] = pk_bf16(wp[2 * HDc], wp[3 * HDc]);
      *(u32x2*)&Wt[n * 64 + k0] = p;
    }
    __syncthreads();

    for (int ks = 0; ks < 2; ++ks) {
      Frag16 a;
      a.u[0] = *(const u32x4*)&Xs[arow * 64 + ks * 32 + half * 8];
      a.u[1] = *(const u32x4*)&Xs[arow * 64 + ks * 32 + 16 + half * 8];
      for (int nt = 0; nt < 4; ++nt) {
        Frag16 bfg;
        const int n = nt * 16 + (lane & 15);
        bfg.u[0] = *(const u32x4*)&Wt[n * 64 + ks * 32 + half * 8];
        bfg.u[1] = *(const u32x4*)&Wt[n * 64 + ks * 32 + 16 + half * 8];
        acc[nt] = __builtin_amdgcn_wmma_f32_16x16x32_bf16(
            false, a.v, false, bfg.v, (short)0, acc[nt], false, false);
      }
    }
  }

  // epilogue: +bias, fp32 -> bf16
  for (int nt = 0; nt < 4; ++nt) {
    const int n   = nt * 16 + (lane & 15);
    const float b = bias[h * HDc + n];
    for (int r = 0; r < 8; ++r) {
      const unsigned int g = (unsigned int)(row0 + wave * 16 + half * 8 + r);
      const unsigned int bb = g >> 11;         // / Sc
      const unsigned int s  = g & (Sc - 1);    // % Sc
      const unsigned short val = f2bf(acc[nt][r] + b);
      if (z == 2)  // V transposed: [b][h][e][s]
        Out[(((size_t)bb * Hc + h) * HDc + n) * Sc + s] = val;
      else         // Q/K: [b][h][s][e]
        Out[(((size_t)bb * Hc + h) * Sc + s) * HDc + n] = val;
    }
  }
}

// ---------------------------------------------------------------------------
// Kernel 2: flash attention per (b, h, 64-query tile)
// grid = (S/64, H, B); block = 128 (4 waves); wave -> 16 query rows
// ---------------------------------------------------------------------------
__global__ __launch_bounds__(128)
void attn_kernel(const unsigned short* __restrict__ Qb,
                 const unsigned short* __restrict__ Kb,
                 const unsigned short* __restrict__ Vb,   // transposed [b][h][e][s]
                 const int* __restrict__ mask,
                 unsigned short* __restrict__ Ob)
{
  __shared__ unsigned short Ks[64 * 64];       // K tile, [key][e]
  __shared__ unsigned short Vs[64 * 64];       // V tile, [e][key]
  __shared__ unsigned short Ps[4 * 16 * 64];   // per-wave P tile, [row][key]

  const int tid  = threadIdx.x;
  const int lane = tid & 31;
  const int wave = tid >> 5;
  const int q0   = blockIdx.x * 64;
  const int h    = blockIdx.y;
  const int bb   = blockIdx.z;
  const size_t headbase = (((size_t)bb * Hc) + h) * Sc * HDc;  // same for V^T
  const float scale = 0.125f;                  // 1/sqrt(64)
  const int half = lane >> 4;

  // Q A-fragments for this wave's 16 rows (HD=64 -> two K=32 steps)
  Frag16 aq[2];
  {
    const int qrow = q0 + wave * 16 + (lane & 15);
    const unsigned short* qp = Qb + headbase + (size_t)qrow * HDc;
    for (int ks = 0; ks < 2; ++ks) {
      aq[ks].u[0] = *(const u32x4*)&qp[ks * 32 + half * 8];
      aq[ks].u[1] = *(const u32x4*)&qp[ks * 32 + 16 + half * 8];
    }
  }

  v8f acc_o[4] = {};
  float m_run[8], l_run[8];
  for (int r = 0; r < 8; ++r) { m_run[r] = -3.0e38f; l_run[r] = 0.f; }

  unsigned short* Pw = Ps + wave * 16 * 64;

  for (int kb = 0; kb < Sc / 64; ++kb) {
    __syncthreads();
    const unsigned short* kbase = Kb + headbase + (size_t)kb * 64 * HDc;
    const unsigned short* vtb   = Vb + headbase + (size_t)kb * 64;  // +e*Sc
    // stage K tile: 64 keys x 64 e, 16B async chunks
    for (int i = tid; i < 512; i += 128) {
      const int key = i >> 3, e8 = (i & 7) * 8;
      copy16_g2l(&Ks[key * 64 + e8], &kbase[key * 64 + e8]);
    }
    // stage V^T tile: 64 e x 64 keys, 16B async chunks
    for (int i = tid; i < 512; i += 128) {
      const int e = i >> 3, k8 = (i & 7) * 8;
      copy16_g2l(&Vs[e * 64 + k8], &vtb[(size_t)e * Sc + k8]);
    }
    async_join();
    __syncthreads();

    // S = Q * K^T  (B-matrix column = key, contiguous e per lane)
    v8f sacc[4] = {};
    for (int kt = 0; kt < 4; ++kt) {
      for (int ks = 0; ks < 2; ++ks) {
        Frag16 bk_;
        const int keycol = kt * 16 + (lane & 15);
        const int eoff   = ks * 32 + half * 8;
        bk_.u[0] = *(const u32x4*)&Ks[keycol * 64 + eoff];
        bk_.u[1] = *(const u32x4*)&Ks[keycol * 64 + eoff + 16];
        sacc[kt] = __builtin_amdgcn_wmma_f32_16x16x32_bf16(
            false, aq[ks].v, false, bk_.v, (short)0, sacc[kt], false, false);
      }
      // scale + key mask (lane owns one key column across its 8 rows)
      const int keyg = kb * 64 + kt * 16 + (lane & 15);
      const int mk   = mask[bb * Sc + keyg];
      for (int r = 0; r < 8; ++r)
        sacc[kt][r] = mk ? sacc[kt][r] * scale : -1e9f;
    }

    // online softmax: reduce across the 16-lane N group (xor masks keep halves)
    for (int r = 0; r < 8; ++r) {
      float rmax = fmaxf(fmaxf(sacc[0][r], sacc[1][r]),
                         fmaxf(sacc[2][r], sacc[3][r]));
      for (int off = 1; off < 16; off <<= 1)
        rmax = fmaxf(rmax, __shfl_xor(rmax, off, 32));
      const float mnew = fmaxf(m_run[r], rmax);
      const float resc = __expf(m_run[r] - mnew);
      float rsum = 0.f;
      for (int kt = 0; kt < 4; ++kt) {
        const float p = __expf(sacc[kt][r] - mnew);
        sacc[kt][r] = p;
        rsum += p;
      }
      for (int off = 1; off < 16; off <<= 1)
        rsum += __shfl_xor(rsum, off, 32);
      m_run[r] = mnew;
      l_run[r] = l_run[r] * resc + rsum;
      for (int nt = 0; nt < 4; ++nt) acc_o[nt][r] *= resc;
    }

    // C-layout -> A-layout relayout of P via per-wave LDS (DS in-order per wave)
    for (int kt = 0; kt < 4; ++kt) {
      const int col = kt * 16 + (lane & 15);
      for (int r = 0; r < 8; ++r)
        Pw[(half * 8 + r) * 64 + col] = f2bf(sacc[kt][r]);
    }
    Frag16 ap[2];
    {
      const int prow = lane & 15;
      for (int ks = 0; ks < 2; ++ks) {
        ap[ks].u[0] = *(const u32x4*)&Pw[prow * 64 + ks * 32 + half * 8];
        ap[ks].u[1] = *(const u32x4*)&Pw[prow * 64 + ks * 32 + 16 + half * 8];
      }
    }

    // O += P * V  (B-matrix column = e, contiguous keys per lane via Vs[e][key])
    for (int nt = 0; nt < 4; ++nt) {
      for (int ks = 0; ks < 2; ++ks) {
        Frag16 bv_;
        const int ecol = nt * 16 + (lane & 15);
        const int koff = ks * 32 + half * 8;
        bv_.u[0] = *(const u32x4*)&Vs[ecol * 64 + koff];
        bv_.u[1] = *(const u32x4*)&Vs[ecol * 64 + koff + 16];
        acc_o[nt] = __builtin_amdgcn_wmma_f32_16x16x32_bf16(
            false, ap[ks].v, false, bv_.v, (short)0, acc_o[nt], false, false);
      }
    }
  }

  // normalize and store bf16 head output [B,H,S,HD]
  for (int nt = 0; nt < 4; ++nt) {
    const int e = nt * 16 + (lane & 15);
    for (int r = 0; r < 8; ++r) {
      const int srow = q0 + wave * 16 + half * 8 + r;
      Ob[headbase + (size_t)srow * HDc + e] = f2bf(acc_o[nt][r] / l_run[r]);
    }
  }
}

// ---------------------------------------------------------------------------
// Kernel 3: output projection  concat(heads) @ Wo + bo  -> fp32
// grid = (BS/128, D/64); block = 256 (8 waves); K block of 64 == one head
// ---------------------------------------------------------------------------
__global__ __launch_bounds__(256)
void out_proj_kernel(const unsigned short* __restrict__ Ob,
                     const float* __restrict__ Wo,
                     const float* __restrict__ bo,
                     float* __restrict__ out)
{
  __shared__ unsigned short Wt[64 * 64];      // Wo tile TRANSPOSED [n][k], 8 KB

  const int tid  = threadIdx.x;
  const int lane = tid & 31;
  const int wave = tid >> 5;
  const int row0 = blockIdx.x * 128;
  const int n0   = blockIdx.y * 64;
  const int half = lane >> 4;

  v8f acc[4] = {};

  const unsigned int g_a = (unsigned int)(row0 + wave * 16 + (lane & 15));
  const unsigned int ab = g_a >> 11, as = g_a & (Sc - 1);

  for (int kb = 0; kb < Dc / 64; ++kb) {    // kb == head index
    __syncthreads();
    for (int i = tid; i < 1024; i += 256) {
      const int n = i & 63, k0 = (i >> 6) * 4;
      const float* wp = &Wo[(size_t)(kb * 64 + k0) * Dc + n0 + n];
      u32x2 p;
      p[0] = pk_bf16(wp[0], wp[Dc]);
      p[1] = pk_bf16(wp[2 * Dc], wp[3 * Dc]);
      *(u32x2*)&Wt[n * 64 + k0] = p;
    }
    __syncthreads();

    // A fragment straight from bf16 head output (head kb, e = 0..63)
    const unsigned short* ap_ = Ob + (((size_t)ab * Hc + kb) * Sc + as) * HDc;
    for (int ks = 0; ks < 2; ++ks) {
      Frag16 a;
      a.u[0] = *(const u32x4*)&ap_[ks * 32 + half * 8];
      a.u[1] = *(const u32x4*)&ap_[ks * 32 + 16 + half * 8];
      for (int nt = 0; nt < 4; ++nt) {
        Frag16 bw;
        const int n = nt * 16 + (lane & 15);
        bw.u[0] = *(const u32x4*)&Wt[n * 64 + ks * 32 + half * 8];
        bw.u[1] = *(const u32x4*)&Wt[n * 64 + ks * 32 + 16 + half * 8];
        acc[nt] = __builtin_amdgcn_wmma_f32_16x16x32_bf16(
            false, a.v, false, bw.v, (short)0, acc[nt], false, false);
      }
    }
  }

  for (int nt = 0; nt < 4; ++nt) {
    const int n    = n0 + nt * 16 + (lane & 15);
    const float bv = bo[n];
    for (int r = 0; r < 8; ++r) {
      const int g = row0 + wave * 16 + half * 8 + r;
      out[(size_t)g * Dc + n] = acc[nt][r] + bv;
    }
  }
}

// ---------------------------------------------------------------------------
// Launch
// ---------------------------------------------------------------------------
extern "C" void kernel_launch(void* const* d_in, const int* in_sizes, int n_in,
                              void* d_out, int out_size, void* d_ws, size_t ws_size,
                              hipStream_t stream) {
  (void)in_sizes; (void)n_in; (void)out_size; (void)ws_size;
  const float* x  = (const float*)d_in[0];
  const int*  msk = (const int*)d_in[1];
  const float* Wq = (const float*)d_in[2];
  const float* bq = (const float*)d_in[3];
  const float* Wk = (const float*)d_in[4];
  const float* bk = (const float*)d_in[5];
  const float* Wv = (const float*)d_in[6];
  const float* bv = (const float*)d_in[7];
  const float* Wo = (const float*)d_in[8];
  const float* bo = (const float*)d_in[9];
  float* out = (float*)d_out;

  // workspace: Q,K,V^T,O_heads in bf16, each B*H*S*HD elems (8 MB) -> 32 MB
  const size_t headElems = (size_t)Bc * Hc * Sc * HDc;
  unsigned short* Qb = (unsigned short*)d_ws;
  unsigned short* Kb = Qb + headElems;
  unsigned short* Vb = Kb + headElems;   // transposed [b][h][e][s]
  unsigned short* Ob = Vb + headElems;

  qkv_proj_kernel<<<dim3((Bc * Sc) / 128, Hc, 3), 256, 0, stream>>>(
      x, Wq, bq, Wk, bk, Wv, bv, Qb, Kb, Vb);
  attn_kernel<<<dim3(Sc / 64, Hc, Bc), 128, 0, stream>>>(Qb, Kb, Vb, msk, Ob);
  out_proj_kernel<<<dim3((Bc * Sc) / 128, Dc / 64), 256, 0, stream>>>(
      Ob, Wo, bo, out);
}
